// TiSASGNN_75290776699105
// MI455X (gfx1250) — matrix operand order repaired and checked by reading, not compiled, not run on hardware
//
#include <hip/hip_runtime.h>

typedef __attribute__((ext_vector_type(16))) _Float16 v16h;
typedef __attribute__((ext_vector_type(8)))  _Float16 v8h;
typedef __attribute__((ext_vector_type(4)))  _Float16 v4h;
typedef __attribute__((ext_vector_type(8)))  float    v8f;
typedef __attribute__((ext_vector_type(4)))  float    v4f;

#define HIDDEN 128
#define LDSK   144   // padded row stride in halves: 288B rows (16B aligned, bank-skewed)

// ---------------- init: agg = 0 (float4 stores), deg = 1 (self loop) --------
__global__ void gcn_init_kernel(float* __restrict__ deg, float* __restrict__ agg,
                                int n_nodes) {
    long long idx = (long long)blockIdx.x * blockDim.x + threadIdx.x;
    long long total4 = (long long)n_nodes * (HIDDEN / 4);
    if (idx < total4) {
        v4f z = {0.0f, 0.0f, 0.0f, 0.0f};
        ((v4f*)agg)[idx] = z;
    }
    if (idx < n_nodes) deg[idx] = 1.0f;
}

// ---------------- degree count over dst ----------------
__global__ void gcn_deg_kernel(const int* __restrict__ dst, float* __restrict__ deg,
                               int n_edges) {
    int e = blockIdx.x * blockDim.x + threadIdx.x;
    if (e < n_edges) atomicAdd(&deg[dst[e]], 1.0f);
}

// ---------------- dinv = rsqrt(deg) ----------------
__global__ void gcn_dinv_kernel(float* __restrict__ deg, int n_nodes) {
    int i = blockIdx.x * blockDim.x + threadIdx.x;
    if (i < n_nodes) {
        float d = deg[i];
        deg[i] = (d > 0.0f) ? rsqrtf(d) : 0.0f;
    }
}

// ---------------- h = x @ W  (f16 WMMA, f32 accumulate) ----------------
// Block: 256 threads = 8 waves. Block computes 16 rows x 128 cols.
// Wave w computes the 16x16 tile at columns [16w, 16w+16).
// D is staged through LDS so global stores are coalesced b128.
__global__ __launch_bounds__(256)
void gcn_gemm_kernel(const float* __restrict__ x, const float* __restrict__ W,
                     _Float16* __restrict__ h, int n_nodes) {
    __shared__ _Float16 Wt[HIDDEN * LDSK]; // Wt[n][k] = W[k][n], f16
    __shared__ _Float16 At[16 * LDSK];     // At[m][k] = x[row0+m][k], f16
    __shared__ _Float16 Dt[16 * HIDDEN];   // staging for coalesced stores

    const int tid  = threadIdx.x;
    const int row0 = blockIdx.x * 16;

    // Load + transpose + convert W (128x128). W is L2-resident (64KB).
    for (int i = tid; i < HIDDEN * HIDDEN; i += 256) {
        int k = i >> 7, n = i & 127;
        Wt[n * LDSK + k] = (_Float16)W[i];
    }
    // Load + convert 16 rows of x.
    for (int i = tid; i < 16 * HIDDEN; i += 256) {
        int m = i >> 7, k = i & 127;
        int r = row0 + m;
        float v = (r < n_nodes) ? x[(long long)r * HIDDEN + k] : 0.0f;
        At[m * LDSK + k] = (_Float16)v;
    }
    __syncthreads();

    const int wave = tid >> 5;
    const int lane = tid & 31;
    const int n0   = wave * 16;
    const int half = lane >> 4;   // 0: lanes 0-15, 1: lanes 16-31
    const int l16  = lane & 15;

    v8f acc = {};
    #pragma unroll
    for (int kc = 0; kc < 4; ++kc) {
        const int kb = kc * 32;
        // A fragment (16x32 f16, ISA layout): lane's row m=l16.
        //   halves 0..7  -> K = kb +      half*8 + 0..7
        //   halves 8..15 -> K = kb + 16 + half*8 + 0..7
        const _Float16* ap = &At[l16 * LDSK + kb + half * 8];
        v8h a_lo = *(const v8h*)ap;
        v8h a_hi = *(const v8h*)(ap + 16);
        v16h a = __builtin_shufflevector(a_lo, a_hi,
                 0,1,2,3,4,5,6,7,8,9,10,11,12,13,14,15);
        // B fragment (32x16 f16, ISA layout): lane's column n = n0+l16.
        //   halves 0..7  -> K = kb +      half*8 + 0..7
        //   halves 8..15 -> K = kb + 16 + half*8 + 0..7
        const _Float16* bp = &Wt[(n0 + l16) * LDSK + kb + half * 8];
        v8h b_lo = *(const v8h*)bp;
        v8h b_hi = *(const v8h*)(bp + 16);
        v16h b = __builtin_shufflevector(b_lo, b_hi,
                 0,1,2,3,4,5,6,7,8,9,10,11,12,13,14,15);

        acc = __builtin_amdgcn_wmma_f32_16x16x32_f16(
                  /*neg_a=*/false, a, /*neg_b=*/false, b,
                  /*c_mod=*/(short)0, acc, /*reuse_a=*/false, /*reuse_b=*/false);
    }

    // D layout: VGPR v -> M = v + 8*half, N = l16 (within tile). Stage to LDS.
    #pragma unroll
    for (int v = 0; v < 8; ++v) {
        int m = v + half * 8;
        Dt[m * HIDDEN + n0 + l16] = (_Float16)acc[v];
    }
    __syncthreads();

    // Coalesced b128 stores: thread t -> row t/16, 8 cols starting (t%16)*8.
    {
        int m = tid >> 4;
        int c = (tid & 15) * 8;
        int r = row0 + m;
        if (r < n_nodes)
            *(v8h*)&h[(long long)r * HIDDEN + c] = *(const v8h*)&Dt[m * HIDDEN + c];
    }
}

// ---------------- edge scatter: agg[dst] += h[src] * dinv[src]*dinv[dst] ----
// One thread per (edge, 4 features): 8B h loads, 4 f32 atomics.
__global__ __launch_bounds__(256)
void gcn_scatter_kernel(const _Float16* __restrict__ h,
                        const int* __restrict__ src, const int* __restrict__ dst,
                        const float* __restrict__ dinv, float* __restrict__ agg,
                        int n_edges) {
    long long idx = (long long)blockIdx.x * blockDim.x + threadIdx.x;
    int e = (int)(idx >> 5);
    int d = (int)(idx & 31) * 4;
    if (e >= n_edges) return;
    int s = src[e], t = dst[e];
    float norm = dinv[s] * dinv[t];
    v4h hv = *(const v4h*)&h[(long long)s * HIDDEN + d];
    float* ap = &agg[(long long)t * HIDDEN + d];
    #pragma unroll
    for (int j = 0; j < 4; ++j)
        atomicAdd(ap + j, (float)hv[j] * norm);
}

// ---------------- finalize: self-loop + bias + relu + residual + LayerNorm --
// One wave32 per row; 4 features per lane; shfl_xor reductions.
__global__ __launch_bounds__(256)
void gcn_final_kernel(const float* __restrict__ x, const _Float16* __restrict__ h,
                      const float* __restrict__ agg, const float* __restrict__ dinv,
                      const float* __restrict__ bias, const float* __restrict__ gamma,
                      const float* __restrict__ beta, float* __restrict__ out,
                      int n_nodes) {
    const int wave = threadIdx.x >> 5;
    const int lane = threadIdx.x & 31;
    const int row  = blockIdx.x * 8 + wave;
    if (row >= n_nodes) return;

    const long long base = (long long)row * HIDDEN + lane * 4;
    const float di = dinv[row];
    const float selfn = di * di;

    v4f ag = *(const v4f*)&agg[base];
    v4h hs = *(const v4h*)&h[base];
    v4f xs = *(const v4f*)&x[base];
    v4f bi = *(const v4f*)&bias[lane * 4];

    float y[4];
    #pragma unroll
    for (int j = 0; j < 4; ++j) {
        float a = ag[j] + (float)hs[j] * selfn + bi[j];
        a = a > 0.0f ? a : 0.0f;                 // relu
        y[j] = a + xs[j];                        // residual
    }

    float s = y[0] + y[1] + y[2] + y[3];
    #pragma unroll
    for (int o = 16; o >= 1; o >>= 1) s += __shfl_xor(s, o, 32);
    const float mu = s * (1.0f / HIDDEN);

    float vs = 0.0f;
    #pragma unroll
    for (int j = 0; j < 4; ++j) { float d = y[j] - mu; vs += d * d; }
    #pragma unroll
    for (int o = 16; o >= 1; o >>= 1) vs += __shfl_xor(vs, o, 32);
    const float inv = rsqrtf(vs * (1.0f / HIDDEN) + 1e-8f);

    v4f ga = *(const v4f*)&gamma[lane * 4];
    v4f be = *(const v4f*)&beta[lane * 4];
    v4f o4;
    #pragma unroll
    for (int j = 0; j < 4; ++j)
        o4[j] = (y[j] - mu) * inv * ga[j] + be[j];
    *(v4f*)&out[base] = o4;
}

extern "C" void kernel_launch(void* const* d_in, const int* in_sizes, int n_in,
                              void* d_out, int out_size, void* d_ws, size_t ws_size,
                              hipStream_t stream) {
    const float* x     = (const float*)d_in[0];
    const int*   edge  = (const int*)d_in[1];
    const float* W     = (const float*)d_in[2];
    const float* bias  = (const float*)d_in[3];
    const float* gamma = (const float*)d_in[4];
    const float* beta  = (const float*)d_in[5];

    const int n_nodes = in_sizes[0] / HIDDEN;
    const int n_edges = in_sizes[1] / 2;
    const int* src = edge;
    const int* dst = edge + n_edges;

    // Workspace carve-up (256B aligned): deg/dinv | h(f16) | agg(f32)
    char* ws = (char*)d_ws;
    float* deg = (float*)ws;
    size_t off = ((size_t)n_nodes * sizeof(float) + 255) & ~(size_t)255;
    _Float16* h = (_Float16*)(ws + off);
    off += ((size_t)n_nodes * HIDDEN * sizeof(_Float16) + 255) & ~(size_t)255;
    float* agg = (float*)(ws + off);

    const long long total4 = (long long)n_nodes * (HIDDEN / 4);

    gcn_init_kernel<<<(int)((total4 + 255) / 256), 256, 0, stream>>>(deg, agg, n_nodes);
    gcn_deg_kernel<<<(n_edges + 255) / 256, 256, 0, stream>>>(dst, deg, n_edges);
    gcn_dinv_kernel<<<(n_nodes + 255) / 256, 256, 0, stream>>>(deg, n_nodes);

    gcn_gemm_kernel<<<(n_nodes + 15) / 16, 256, 0, stream>>>(x, W, h, n_nodes);

    const long long ework = (long long)n_edges * 32;
    gcn_scatter_kernel<<<(int)((ework + 255) / 256), 256, 0, stream>>>(
        h, src, dst, deg, agg, n_edges);

    gcn_final_kernel<<<(n_nodes + 7) / 8, 256, 0, stream>>>(
        x, h, agg, deg, bias, gamma, beta, (float*)d_out, n_nodes);
}
